// GGNN2_77764677862202
// MI455X (gfx1250) — matrix operand order repaired and checked by reading, not compiled, not run on hardware
//
#include <hip/hip_runtime.h>
#include <hip/hip_bf16.h>
#include <math.h>

// ---------------------------------------------------------------------------
// GGNN (2 layers) on MI455X / gfx1250.
//  - Dense GEMMs: v_wmma_f32_16x16x32_f16 (f16 in, f32 accum), weights
//    pre-packed into fragment order once.
//  - Scatter-aggregate (roofline-dominant, ~1.3GB L2 traffic): coalesced
//    global_atomic_add_f32 with wave-uniform (scalar) edge-index loads.
//  - GRU: async LDS staging (global_load_async_to_lds_b128 + s_wait_asynccnt)
//    feeding WMMA fragments, gates fused in LDS, h updated in place.
// ---------------------------------------------------------------------------

constexpr int kN  = 100000;   // nodes
constexpr int kE  = 1600000;  // edges
constexpr int kG  = 1000;     // graphs
constexpr int kD  = 50;       // feature dim
constexpr int kDP = 64;       // padded feature dim (multiple of 32 for WMMA K)
constexpr int kTD = 150;      // 3*D (GRU gates)
constexpr int kTDP = 160;     // padded 3*D
constexpr int kC  = 6;        // classes
constexpr int kMT = kN / 16;  // 6250 M-tiles

typedef __attribute__((ext_vector_type(16))) _Float16 v16h;
typedef __attribute__((ext_vector_type(8)))  float    v8f;

// Fragment (ntile,kstep,lane,half) -> (k,n) for 16x16x32 f16 B-matrix layout:
// g = lane>>4; halves 0..7 -> K = kstep*32 + 8g + half, halves 8..15 ->
// K = kstep*32 + 16 + 8g + (half-8); N = ntile*16 + (lane&15).
__device__ __forceinline__ void frag_kn(int ntile, int kstep, int lane, int half,
                                        int* k, int* n) {
    *n = ntile * 16 + (lane & 15);
    *k = kstep * 32 + ((lane >> 4) << 3) + (half & 7) + ((half >> 3) << 4);
}

// A-matrix fragment (16x32 f16): lane row M = lane&15; group g = lane>>4;
// halves 0..7 -> K = kk + 8g + j ; halves 8..15 -> K = kk + 8g + 16 + (j-8).
__device__ __forceinline__ v16h load_a_frag(const float* __restrict__ rowp,
                                            int kk, int lane, float scale) {
    const int g = (lane >> 4) & 1;
    const float4* p0 = (const float4*)(rowp + kk + 8 * g);
    const float4* p1 = (const float4*)(rowp + kk + 8 * g + 16);
    float4 f0 = p0[0], f1 = p0[1];
    float4 f2 = p1[0], f3 = p1[1];
    v16h a;
    a[0]  = (_Float16)(f0.x * scale); a[1]  = (_Float16)(f0.y * scale);
    a[2]  = (_Float16)(f0.z * scale); a[3]  = (_Float16)(f0.w * scale);
    a[4]  = (_Float16)(f1.x * scale); a[5]  = (_Float16)(f1.y * scale);
    a[6]  = (_Float16)(f1.z * scale); a[7]  = (_Float16)(f1.w * scale);
    a[8]  = (_Float16)(f2.x * scale); a[9]  = (_Float16)(f2.y * scale);
    a[10] = (_Float16)(f2.z * scale); a[11] = (_Float16)(f2.w * scale);
    a[12] = (_Float16)(f3.x * scale); a[13] = (_Float16)(f3.y * scale);
    a[14] = (_Float16)(f3.z * scale); a[15] = (_Float16)(f3.w * scale);
    return a;
}

// C/D (16x16 f32): lanes 0-15: M = r (VGPR idx), lanes 16-31: M = 8+r; N = lane&15.
__device__ __forceinline__ void store_tile(float* __restrict__ dst, int ld,
                                           int row0, int col0, int lane, v8f c) {
    const int mb  = row0 + ((lane >> 4) << 3);
    const int col = col0 + (lane & 15);
    #pragma unroll
    for (int r = 0; r < 8; ++r) dst[(size_t)(mb + r) * ld + col] = c[r];
}

// Low 32 bits of a flat pointer to __shared__ = workgroup-relative LDS offset.
__device__ __forceinline__ unsigned lds_off(const void* p) {
    return (unsigned)(unsigned long long)p;
}

// gfx1250 async LDS copy: 16B per lane, ASYNCcnt-tracked (inline asm: the
// clang builtin arity differs between toolchains; asm is portable).
__device__ __forceinline__ void async_copy_b128(unsigned lds_byte_off,
                                                const void* gaddr) {
    unsigned long long ga = (unsigned long long)gaddr;
    asm volatile("global_load_async_to_lds_b128 %0, %1, off"
                 :: "v"(lds_byte_off), "v"(ga) : "memory");
}
__device__ __forceinline__ void wait_asynccnt0() {
    asm volatile("s_wait_asynccnt 0x0" ::: "memory");
}

// ---------------------------------------------------------------------------
// One-time weight conversion into WMMA-fragment-ordered f16 buffers.
//   wlfrag : [layer(2)][ntile(4)][kstep(2)][lane(32)][half(16)]  (h @ W)
//   wihfrag: [ntile(10)][kstep(2)][lane(32)][half(16)]           (agg @ w_ih^T)
//   whhfrag: same shape                                          (h   @ w_hh^T)
// ---------------------------------------------------------------------------
__global__ void convert_weights(const float* __restrict__ gw,
                                const float* __restrict__ wih,
                                const float* __restrict__ whh,
                                _Float16* __restrict__ wlfrag,
                                _Float16* __restrict__ wihfrag,
                                _Float16* __restrict__ whhfrag) {
    int t = blockIdx.x * blockDim.x + threadIdx.x;
    if (t < 8192) {                                   // ggnn_weight, both layers
        int layer = t >> 12;
        int rem   = t & 4095;
        int ntile = rem >> 10, kstep = (rem >> 9) & 1;
        int lane  = (rem >> 4) & 31, half = rem & 15;
        int k, n; frag_kn(ntile, kstep, lane, half, &k, &n);
        float v = (k < kD && n < kD) ? gw[layer * kD * kD + k * kD + n] : 0.f;
        wlfrag[t] = (_Float16)v;
    } else if (t < 8192 + 10240) {                    // w_ih^T  (B[k][n] = w_ih[n][k])
        int t2 = t - 8192;
        int ntile = t2 >> 10, kstep = (t2 >> 9) & 1;
        int lane  = (t2 >> 4) & 31, half = t2 & 15;
        int k, n; frag_kn(ntile, kstep, lane, half, &k, &n);
        float v = (k < kD && n < kTD) ? wih[n * kD + k] : 0.f;
        wihfrag[t2] = (_Float16)v;
    } else if (t < 8192 + 20480) {                    // w_hh^T
        int t2 = t - 8192 - 10240;
        int ntile = t2 >> 10, kstep = (t2 >> 9) & 1;
        int lane  = (t2 >> 4) & 31, half = t2 & 15;
        int k, n; frag_kn(ntile, kstep, lane, half, &k, &n);
        float v = (k < kD && n < kTD) ? whh[n * kD + k] : 0.f;
        whhfrag[t2] = (_Float16)v;
    }
}

// h[i][0:50] = x[i][0:50], pad 50:64 with zeros (keeps WMMA K-padding exact).
__global__ void pad_x(const float* __restrict__ x, float* __restrict__ h) {
    long long gid = (long long)blockIdx.x * blockDim.x + threadIdx.x;
    int i = (int)(gid >> 6), j = (int)(gid & 63);
    if (i < kN) h[gid] = (j < kD) ? x[(size_t)i * kD + j] : 0.f;
}

__global__ void deg_kernel(const int* __restrict__ dst, float* __restrict__ deg) {
    int e = blockIdx.x * blockDim.x + threadIdx.x;
    if (e < kE) atomicAdd(&deg[dst[e]], 1.0f);
}

__global__ void invdeg_kernel(float* __restrict__ deg) {
    int i = blockIdx.x * blockDim.x + threadIdx.x;
    if (i < kN) deg[i] = 1.0f / fmaxf(deg[i], 1.0f);
}

__global__ void gcnt_kernel(const int* __restrict__ batch, float* __restrict__ gcnt) {
    int i = blockIdx.x * blockDim.x + threadIdx.x;
    if (i < kN) atomicAdd(&gcnt[batch[i]], 1.0f);
}

// ---------------------------------------------------------------------------
// m = h @ W[layer].  One wave per 16-row strip: A fragments loaded ONCE and
// reused across all four 16-col tiles (8 WMMAs/wave, zero redundant A traffic).
// ---------------------------------------------------------------------------
__global__ void __launch_bounds__(256) gemm_m(const float* __restrict__ h,
                                              const _Float16* __restrict__ bfrag,
                                              float* __restrict__ m) {
    const int wid  = threadIdx.x >> 5;
    const int lane = threadIdx.x & 31;
    const int mtile = blockIdx.x * 8 + wid;
    if (mtile >= kMT) return;                 // wave-uniform: EXEC stays all-1s
    const int row0 = mtile * 16;
    const v16h* bf = (const v16h*)bfrag;
    const float* rowp = h + (size_t)(row0 + (lane & 15)) * kDP;
    v16h a0 = load_a_frag(rowp, 0,  lane, 1.0f);
    v16h a1 = load_a_frag(rowp, 32, lane, 1.0f);
    #pragma unroll
    for (int nt = 0; nt < 4; ++nt) {
        v8f c = {};
        c = __builtin_amdgcn_wmma_f32_16x16x32_f16(false, a0, false,
                bf[(nt * 2 + 0) * 32 + lane], (short)0, c, false, false);
        c = __builtin_amdgcn_wmma_f32_16x16x32_f16(false, a1, false,
                bf[(nt * 2 + 1) * 32 + lane], (short)0, c, false, false);
        store_tile(m, kDP, row0, nt * 16, lane, c);
    }
}

// ---------------------------------------------------------------------------
// agg[dst] += m[src].  32 threads per edge -> wave-uniform edge index, forced
// into SGPRs via readfirstlane so src/dst become scalar loads; lane j handles
// features j and (j+32 if < 50): zero wasted atomic slots, coalesced payload.
// ---------------------------------------------------------------------------
__global__ void scatter_edges(const int* __restrict__ src, const int* __restrict__ dst,
                              const float* __restrict__ m, float* __restrict__ agg) {
    long long gid = (long long)blockIdx.x * blockDim.x + threadIdx.x;
    int e = (int)(gid >> 5);
    e = __builtin_amdgcn_readfirstlane(e);    // wave-uniform -> scalar index loads
    int j = threadIdx.x & 31;
    if (e < kE) {
        int s = src[e], d = dst[e];
        const float* mrow = m + (size_t)s * kDP;
        float* arow = agg + (size_t)d * kDP;
        atomicAdd(&arow[j], mrow[j]);
        if (j < kD - 32)                      // j < 18 -> features 32..49
            atomicAdd(&arow[j + 32], mrow[j + 32]);
    }
}

// ---------------------------------------------------------------------------
// Fused GRU. Block = 16 rows, 10 waves.
//   Stage 0: async-copy the block's 16 rows of agg and h (8KB) into LDS
//            (global_load_async_to_lds_b128, ASYNCcnt), wait + barrier.
//   Stage 1: wave w computes 16-col tile w of gi = (agg/deg)@w_ih^T and of
//            gh = h@w_hh^T from the LDS copies (4 WMMAs/wave), parks in LDS.
//   Stage 2: gates + in-place h update (h_old read from LDS stage).
// ---------------------------------------------------------------------------
__global__ void __launch_bounds__(320) gru_fused(const float* __restrict__ agg,
                                                 const float* __restrict__ invdeg,
                                                 float* __restrict__ h,
                                                 const _Float16* __restrict__ wihfrag,
                                                 const _Float16* __restrict__ whhfrag,
                                                 const float* __restrict__ b_ih,
                                                 const float* __restrict__ b_hh) {
    __shared__ __align__(16) float s_agg[16][kDP];   // 4KB
    __shared__ __align__(16) float s_h[16][kDP];     // 4KB
    __shared__ float gi_lds[16][kTDP];               // 10KB
    __shared__ float gh_lds[16][kTDP];               // 10KB
    const int wid  = threadIdx.x >> 5;      // 0..9 == ntile
    const int lane = threadIdx.x & 31;
    const int row0 = blockIdx.x * 16;
    const int r = lane & 15;                // row within tile owned by this lane

    // ---- stage 0: async global->LDS (16B/lane, 256 chunks per 4KB matrix) ----
    {
        int t = threadIdx.x;
        if (t < 256) {                       // waves 0..7 (uniform per wave)
            async_copy_b128(lds_off(&s_agg[0][0]) + (unsigned)t * 16u,
                            agg + (size_t)row0 * kDP + t * 4);
            async_copy_b128(lds_off(&s_h[0][0]) + (unsigned)t * 16u,
                            h + (size_t)row0 * kDP + t * 4);
        }
    }
    const float sc = invdeg[row0 + r];      // fold mean-divide into A fragment
    wait_asynccnt0();
    __syncthreads();

    // ---- stage 1: WMMA from LDS ----
    const v16h* bih = (const v16h*)wihfrag;
    const v16h* bhh = (const v16h*)whhfrag;
    v8f cgi = {}, cgh = {};
    #pragma unroll
    for (int ks = 0; ks < 2; ++ks) {
        v16h ai = load_a_frag(&s_agg[r][0], ks * 32, lane, sc);
        cgi = __builtin_amdgcn_wmma_f32_16x16x32_f16(false, ai, false,
                bih[(wid * 2 + ks) * 32 + lane], (short)0, cgi, false, false);
        v16h ah = load_a_frag(&s_h[r][0], ks * 32, lane, 1.0f);
        cgh = __builtin_amdgcn_wmma_f32_16x16x32_f16(false, ah, false,
                bhh[(wid * 2 + ks) * 32 + lane], (short)0, cgh, false, false);
    }
    const int mb  = (lane >> 4) << 3;
    const int col = wid * 16 + (lane & 15);
    #pragma unroll
    for (int rr = 0; rr < 8; ++rr) {
        gi_lds[mb + rr][col] = cgi[rr];
        gh_lds[mb + rr][col] = cgh[rr];
    }
    __syncthreads();

    // ---- stage 2: gates, h updated in place ----
    for (int idx = threadIdx.x; idx < 16 * kD; idx += 320) {
        int rr = idx / kD, cc = idx % kD;
        float ir  = gi_lds[rr][cc]           + b_ih[cc];
        float iz  = gi_lds[rr][kD + cc]      + b_ih[kD + cc];
        float in_ = gi_lds[rr][2 * kD + cc]  + b_ih[2 * kD + cc];
        float hr  = gh_lds[rr][cc]           + b_hh[cc];
        float hz  = gh_lds[rr][kD + cc]      + b_hh[kD + cc];
        float hn  = gh_lds[rr][2 * kD + cc]  + b_hh[2 * kD + cc];
        float rg = 1.f / (1.f + __expf(-(ir + hr)));
        float zg = 1.f / (1.f + __expf(-(iz + hz)));
        float ng = tanhf(in_ + rg * hn);
        float ho = s_h[rr][cc];
        h[(size_t)(row0 + rr) * kDP + cc] = (1.f - zg) * ng + zg * ho;
    }
}

// pooled[batch[i]] += relu(h[i])
__global__ void pool_kernel(const float* __restrict__ h, const int* __restrict__ batch,
                            float* __restrict__ pooled) {
    long long gid = (long long)blockIdx.x * blockDim.x + threadIdx.x;
    int i = (int)(gid >> 6), j = (int)(gid & 63);
    if (i < kN && j < kD)
        atomicAdd(&pooled[(size_t)batch[i] * kDP + j], fmaxf(h[gid], 0.f));
}

// logits = (pooled/cnt) @ fc_w^T + fc_b ; log_softmax.  1000x6x50 MACs: trivial.
__global__ void final_kernel(const float* __restrict__ pooled,
                             const float* __restrict__ gcnt,
                             const float* __restrict__ fcw,
                             const float* __restrict__ fcb,
                             float* __restrict__ out) {
    int g = blockIdx.x * blockDim.x + threadIdx.x;
    if (g >= kG) return;
    float inv = 1.f / fmaxf(gcnt[g], 1.f);
    float logits[kC];
    #pragma unroll
    for (int j = 0; j < kC; ++j) {
        float acc = fcb[j];
        for (int c = 0; c < kD; ++c)
            acc += pooled[(size_t)g * kDP + c] * inv * fcw[j * kD + c];
        logits[j] = acc;
    }
    float mx = logits[0];
    #pragma unroll
    for (int j = 1; j < kC; ++j) mx = fmaxf(mx, logits[j]);
    float s = 0.f;
    #pragma unroll
    for (int j = 0; j < kC; ++j) s += __expf(logits[j] - mx);
    float lse = mx + __logf(s);
    #pragma unroll
    for (int j = 0; j < kC; ++j) out[g * kC + j] = logits[j] - lse;
}

extern "C" void kernel_launch(void* const* d_in, const int* in_sizes, int n_in,
                              void* d_out, int out_size, void* d_ws, size_t ws_size,
                              hipStream_t stream) {
    (void)in_sizes; (void)n_in; (void)out_size; (void)ws_size;
    const float* x    = (const float*)d_in[0];
    const int*   ei   = (const int*)d_in[1];
    const int*   batch= (const int*)d_in[2];
    const float* gw   = (const float*)d_in[3];
    const float* wih  = (const float*)d_in[4];
    const float* whh  = (const float*)d_in[5];
    const float* bih  = (const float*)d_in[6];
    const float* bhh  = (const float*)d_in[7];
    const float* fcw  = (const float*)d_in[8];
    const float* fcb  = (const float*)d_in[9];
    float* out = (float*)d_out;
    const int* src = ei;
    const int* dst = ei + kE;

    // ---- carve workspace (~78 MB) ----
    char* ws = (char*)d_ws;
    size_t off = 0;
    auto carve = [&](size_t bytes) {
        char* p = ws + off;
        off = (off + bytes + 255) & ~(size_t)255;
        return p;
    };
    float* h      = (float*)carve((size_t)kN * kDP * 4);
    float* m      = (float*)carve((size_t)kN * kDP * 4);
    float* agg    = (float*)carve((size_t)kN * kDP * 4);
    float* invdeg = (float*)carve((size_t)kN * 4);
    float* gcnt   = (float*)carve((size_t)kG * 4);
    float* pooled = (float*)carve((size_t)kG * kDP * 4);
    _Float16* wlfrag  = (_Float16*)carve(8192 * 2);
    _Float16* wihfrag = (_Float16*)carve(10240 * 2);
    _Float16* whhfrag = (_Float16*)carve(10240 * 2);

    hipMemsetAsync(invdeg, 0, (size_t)kN * 4, stream);
    hipMemsetAsync(gcnt,   0, (size_t)kG * 4, stream);
    hipMemsetAsync(pooled, 0, (size_t)kG * kDP * 4, stream);

    convert_weights<<<112, 256, 0, stream>>>(gw, wih, whh, wlfrag, wihfrag, whhfrag);
    pad_x<<<(kN * kDP) / 256, 256, 0, stream>>>(x, h);            // 25000 blocks
    deg_kernel<<<kE / 256, 256, 0, stream>>>(dst, invdeg);        // 6250 blocks
    invdeg_kernel<<<(kN + 255) / 256, 256, 0, stream>>>(invdeg);
    gcnt_kernel<<<(kN + 255) / 256, 256, 0, stream>>>(batch, gcnt);

    for (int layer = 0; layer < 2; ++layer) {
        gemm_m<<<(kMT + 7) / 8, 256, 0, stream>>>(h, wlfrag + layer * 4096, m);
        hipMemsetAsync(agg, 0, (size_t)kN * kDP * 4, stream);
        scatter_edges<<<(int)(((long long)kE * 32) / 256), 256, 0, stream>>>(src, dst, m, agg);
        gru_fused<<<kN / 16, 320, 0, stream>>>(agg, invdeg, h, wihfrag, whhfrag, bih, bhh);
    }

    pool_kernel<<<(kN * kDP) / 256, 256, 0, stream>>>(h, batch, pooled);
    final_kernel<<<(kG + 255) / 256, 256, 0, stream>>>(pooled, gcnt, fcw, fcb, out);
}
